// LstmAggregator_6854767804437
// MI455X (gfx1250) — matrix operand order repaired and checked by reading, not compiled, not run on hardware
//
#include <hip/hip_runtime.h>
#include <hip/hip_bf16.h>

typedef __bf16 bf16_t;
typedef __attribute__((ext_vector_type(8)))  bf16_t v8bf;
typedef __attribute__((ext_vector_type(16))) bf16_t v16bf;
typedef __attribute__((ext_vector_type(8)))  float  v8f;
typedef int v4i_t __attribute__((vector_size(16)));

#define DIN   128
#define HID   128
#define DOUT  128
#define GATES 512            // 4*HID
#define MAXDEG 48

// ---- gfx1250 async global->LDS path (guarded; sync fallback) ----
#if defined(__has_builtin)
#if __has_builtin(__builtin_amdgcn_global_load_async_to_lds_b128) && \
    __has_builtin(__builtin_amdgcn_s_wait_asynccnt)
#define USE_ASYNC_LDS 1
#endif
#endif
#ifndef USE_ASYNC_LDS
#define USE_ASYNC_LDS 0
#endif

#if USE_ASYNC_LDS
typedef __attribute__((address_space(1))) v4i_t as1_v4i;
typedef __attribute__((address_space(3))) v4i_t as3_v4i;
#endif

__device__ __forceinline__ void copy16_g2l(bf16_t* ldst, const bf16_t* gsrc) {
#if USE_ASYNC_LDS
    __builtin_amdgcn_global_load_async_to_lds_b128(
        (as1_v4i*)gsrc, (as3_v4i*)ldst, 0, 0);
#else
    *(v8bf*)ldst = *(const v8bf*)gsrc;
#endif
}
__device__ __forceinline__ void wait_async_copies() {
#if USE_ASYNC_LDS
    __builtin_amdgcn_s_wait_asynccnt(0);
#endif
}

// ---- cheap transcendentals: v_exp_f32 + v_rcp_f32, co-execute with WMMA ----
__device__ __forceinline__ float fast_rcp(float x) {
#if defined(__has_builtin)
#if __has_builtin(__builtin_amdgcn_rcpf)
    return __builtin_amdgcn_rcpf(x);
#else
    return 1.0f / x;
#endif
#else
    return 1.0f / x;
#endif
}
__device__ __forceinline__ float sigf(float x) {
    return fast_rcp(1.0f + __expf(-x));
}
__device__ __forceinline__ float tanh_fast(float x) {
    return 1.0f - 2.0f * fast_rcp(__expf(2.0f * x) + 1.0f);
}

// Build a v16bf A/B fragment from two 16-byte chunks.
__device__ __forceinline__ v16bf ld16(const bf16_t* p0, const bf16_t* p1) {
    v8bf lo = *(const v8bf*)p0;
    v8bf hi = *(const v8bf*)p1;
    return __builtin_shufflevector(lo, hi, 0, 1, 2, 3, 4, 5, 6, 7,
                                           8, 9, 10, 11, 12, 13, 14, 15);
}

#define WMMA_BF16(A, B, C) \
    __builtin_amdgcn_wmma_f32_16x16x32_bf16(false, (A), false, (B), (short)0, (C), false, false)

// ---------------------------------------------------------------------------
// Prep kernel 1: starts[n] = lower_bound(edge_src, n), n in [0, N]
// ---------------------------------------------------------------------------
__global__ void prep_starts(const int* __restrict__ esrc, int* __restrict__ starts,
                            int N, int E) {
    int n = blockIdx.x * blockDim.x + threadIdx.x;
    if (n > N) return;
    int lo = 0, hi = E;
    while (lo < hi) {
        int mid = (lo + hi) >> 1;
        if (esrc[mid] < n) lo = mid + 1; else hi = mid;
    }
    starts[n] = lo;
}

// ---------------------------------------------------------------------------
// Prep kernel 2: weights f32 -> bf16 (+ W_out transpose, bias sum)
// ---------------------------------------------------------------------------
__global__ void prep_weights(const float* __restrict__ W_ih,
                             const float* __restrict__ W_hh,
                             const float* __restrict__ b_ih,
                             const float* __restrict__ b_hh,
                             const float* __restrict__ W_out,
                             bf16_t* __restrict__ gwih,
                             bf16_t* __restrict__ gwhh,
                             bf16_t* __restrict__ gwT,
                             float*  __restrict__ gbias) {
    int i = blockIdx.x * blockDim.x + threadIdx.x;
    if (i < GATES * DIN) gwih[i] = (bf16_t)W_ih[i];
    if (i < GATES * HID) gwhh[i] = (bf16_t)W_hh[i];
    if (i < DOUT * (DIN + HID)) {
        int n = i / (DIN + HID);          // output column
        int k = i % (DIN + HID);          // concat-input row
        gwT[i] = (bf16_t)W_out[k * DOUT + n];
    }
    if (i < GATES) gbias[i] = b_ih[i] + b_hh[i];
}

// ---------------------------------------------------------------------------
// Prep kernel 3: input_matrix f32 -> bf16 (halves gather HBM traffic)
// ---------------------------------------------------------------------------
__global__ void prep_xbf(const float* __restrict__ X, bf16_t* __restrict__ Xbf,
                         int count) {
    int i = blockIdx.x * blockDim.x + threadIdx.x;
    if (i < count) Xbf[i] = (bf16_t)X[i];
}

// ---------------------------------------------------------------------------
// Main kernel: 256 threads = 8 wave32; one workgroup per 16-node tile.
// Wave w owns hidden slice j in [16w,16w+16) for all 4 gates.
// Software-pipelined: async gather of X_{t+1} overlaps WMMAs of step t;
// single barrier per timestep.
// ---------------------------------------------------------------------------
__global__ __launch_bounds__(256, 1)
void lstm_agg_main(const bf16_t* __restrict__ Xbf,
                   const int*   __restrict__ etrg,
                   const int*   __restrict__ starts,
                   const bf16_t* __restrict__ gwih,
                   const bf16_t* __restrict__ gwhh,
                   const bf16_t* __restrict__ gwT,
                   const float*  __restrict__ gbias,
                   float* __restrict__ out,
                   int N, int numTiles) {
    __shared__ bf16_t sXT[2][16][DIN];   // double-buffered neighbor features
    __shared__ bf16_t sH[2][16][HID];    // ping-pong hidden state
    __shared__ bf16_t sXN[16][DIN];      // node's own features
    __shared__ int    sStart[16];
    __shared__ int    sDeg[16];

    const int tid  = threadIdx.x;
    const int wv   = tid >> 5;           // wave id 0..7 -> hidden slice
    const int lane = tid & 31;
    const int half = lane >> 4;          // K-half selector for fragments
    const int nn   = lane & 15;          // M (A) / N (B, C/D) index

    // ---- preload per-wave weight slices + bias into VGPRs ----
    v16bf wih[4][4], whh[4][4];
    v8f   bvec[4];
#pragma unroll
    for (int tg = 0; tg < 4; ++tg) {     // gate: 0=i 1=f 2=g 3=o
        const int R = tg * HID + wv * 16 + nn;
        float b = gbias[R];
#pragma unroll
        for (int v = 0; v < 8; ++v) bvec[tg][v] = b;
#pragma unroll
        for (int kc = 0; kc < 4; ++kc) {
            const bf16_t* p = gwih + (size_t)R * DIN + kc * 32 + 16 * half;
            wih[tg][kc] = ld16(p, p + 8);
            const bf16_t* q = gwhh + (size_t)R * HID + kc * 32 + 16 * half;
            whh[tg][kc] = ld16(q, q + 8);
        }
    }

    const int gm = tid >> 4;             // gather row 0..15
    const int gc = (tid & 15) * 8;       // gather col base (8 bf16 / thread)

    for (int tile = blockIdx.x; tile < numTiles; tile += gridDim.x) {
        const int nodeBase = tile * 16;

        if (tid < 16) {
            int node = nodeBase + tid;
            int s = 0, e = 0;
            if (node < N) { s = starts[node]; e = starts[node + 1]; }
            int d = e - s;
            if (d > MAXDEG) d = MAXDEG;
            if (d < 0) d = 0;
            sStart[tid] = s;
            sDeg[tid]   = d;
        }
        {   // stage node's own features + zero H[0]
            int node = nodeBase + gm;
            if (node < N) {
                copy16_g2l(&sXN[gm][gc], Xbf + (size_t)node * DIN + gc);
            } else {
#pragma unroll
                for (int j = 0; j < 8; ++j) sXN[gm][gc + j] = (bf16_t)0.0f;
            }
#pragma unroll
            for (int j = 0; j < 8; ++j) sH[0][gm][gc + j] = (bf16_t)0.0f;
        }
        wait_async_copies();
        __syncthreads();      // sStart/sDeg visible; sXN/H0 staged

        int maxd = 0;
#pragma unroll
        for (int m = 0; m < 16; ++m) maxd = max(maxd, sDeg[m]);

        int dgreg[8];
#pragma unroll
        for (int v = 0; v < 8; ++v) dgreg[v] = sDeg[8 * half + v];

        float creg[8], hreg[8];
#pragma unroll
        for (int v = 0; v < 8; ++v) { creg[v] = 0.0f; hreg[v] = 0.0f; }

        const int gs = sStart[gm];
        const int gd = sDeg[gm];

        // prologue: async-gather X_0 into sXT[0]
        if (0 < gd) {
            int srcn = etrg[gs];
            copy16_g2l(&sXT[0][gm][gc], Xbf + (size_t)srcn * DIN + gc);
        }
        wait_async_copies();
        __syncthreads();

        for (int t = 0; t < maxd; ++t) {
            // ---- issue async gather of X_{t+1}; completes under this step's
            //      WMMAs, waited at end-of-step barrier ----
            if (t + 1 < gd) {
                int srcn = etrg[gs + t + 1];
                copy16_g2l(&sXT[(t + 1) & 1][gm][gc],
                           Xbf + (size_t)srcn * DIN + gc);
                if (t + 2 < gd) {
                    int s2 = etrg[gs + t + 2];
                    __builtin_prefetch(Xbf + (size_t)s2 * DIN + gc, 0, 1);
                }
            }

            const bf16_t* xcur  = &sXT[t & 1][0][0];
            const bf16_t* hcur  = &sH[t & 1][0][0];
            bf16_t*       hnext = &sH[(t + 1) & 1][0][0];

            // ---- gates = X_t @ Wih^T + H @ Whh^T + b  (32 WMMAs/wave) ----
            v16bf ax[4], ah[4];
#pragma unroll
            for (int kc = 0; kc < 4; ++kc) {
                const bf16_t* ar = xcur + nn * DIN + kc * 32 + 16 * half;
                ax[kc] = ld16(ar, ar + 8);
                const bf16_t* hr = hcur + nn * HID + kc * 32 + 16 * half;
                ah[kc] = ld16(hr, hr + 8);
            }

            v8f acc[4];
#pragma unroll
            for (int tg = 0; tg < 4; ++tg)
                acc[tg] = WMMA_BF16(ax[0], wih[tg][0], bvec[tg]);
#pragma unroll
            for (int kc = 1; kc < 4; ++kc)
#pragma unroll
                for (int tg = 0; tg < 4; ++tg)
                    acc[tg] = WMMA_BF16(ax[kc], wih[tg][kc], acc[tg]);
#pragma unroll
            for (int kc = 0; kc < 4; ++kc)
#pragma unroll
                for (int tg = 0; tg < 4; ++tg)
                    acc[tg] = WMMA_BF16(ah[kc], whh[tg][kc], acc[tg]);

            // ---- LSTM cell update, masked for ragged degrees ----
#pragma unroll
            for (int v = 0; v < 8; ++v) {
                float iv = sigf(acc[0][v]);
                float fv = sigf(acc[1][v]);
                float gv = tanh_fast(acc[2][v]);
                float ov = sigf(acc[3][v]);
                float cn = fv * creg[v] + iv * gv;
                float hn = ov * tanh_fast(cn);
                bool keep = (t < dgreg[v]);
                creg[v] = keep ? cn : creg[v];
                hreg[v] = keep ? hn : hreg[v];
                hnext[(v + 8 * half) * HID + wv * 16 + nn] = (bf16_t)hreg[v];
            }

            // one barrier/step: covers H publish + X_{t+1} async landing
            wait_async_copies();
            __syncthreads();
        }

        // ---- output: out(16x128) = [x | h](16x256) @ W_out(256x128) ----
        const bf16_t* hfin = &sH[maxd & 1][0][0];
        v8f oacc;
#pragma unroll
        for (int v = 0; v < 8; ++v) oacc[v] = 0.0f;
#pragma unroll
        for (int kc = 0; kc < 8; ++kc) {
            const bf16_t* wp = gwT + (size_t)(wv * 16 + nn) * (DIN + HID)
                                   + kc * 32 + 16 * half;
            v16bf wb = ld16(wp, wp + 8);
            const bf16_t* ar = (kc < 4)
                ? &sXN[nn][kc * 32 + 16 * half]
                : hfin + nn * HID + (kc - 4) * 32 + 16 * half;
            v16bf aa = ld16(ar, ar + 8);
            oacc = WMMA_BF16(aa, wb, oacc);
        }
#pragma unroll
        for (int v = 0; v < 8; ++v) {
            int node = nodeBase + v + 8 * half;
            if (node < N) out[(size_t)node * DOUT + wv * 16 + nn] = oacc[v];
        }
        __syncthreads();   // protect shared buffers before next tile
    }
}

// ---------------------------------------------------------------------------
extern "C" void kernel_launch(void* const* d_in, const int* in_sizes, int n_in,
                              void* d_out, int out_size, void* d_ws, size_t ws_size,
                              hipStream_t stream) {
    (void)n_in; (void)out_size; (void)ws_size;
    const float* X     = (const float*)d_in[0];
    const float* W_ih  = (const float*)d_in[1];
    const float* W_hh  = (const float*)d_in[2];
    const float* b_ih  = (const float*)d_in[3];
    const float* b_hh  = (const float*)d_in[4];
    const float* W_out = (const float*)d_in[5];
    const int*   esrc  = (const int*)d_in[6];
    const int*   etrg  = (const int*)d_in[7];

    const int N  = in_sizes[0] / DIN;
    const int E  = in_sizes[6];
    const int ND = N * DIN;

    // Workspace carve-out (256B aligned slices)
    size_t off = 0;
    auto take = [&](size_t bytes) -> char* {
        char* p = (char*)d_ws + off;
        off += (bytes + 255) & ~(size_t)255;
        return p;
    };
    int*    starts = (int*)   take(sizeof(int)    * (size_t)(N + 1));
    bf16_t* gwih   = (bf16_t*)take(sizeof(bf16_t) * (size_t)GATES * DIN);
    bf16_t* gwhh   = (bf16_t*)take(sizeof(bf16_t) * (size_t)GATES * HID);
    bf16_t* gwT    = (bf16_t*)take(sizeof(bf16_t) * (size_t)DOUT * (DIN + HID));
    float*  gbias  = (float*) take(sizeof(float)  * (size_t)GATES);
    bf16_t* Xbf    = (bf16_t*)take(sizeof(bf16_t) * (size_t)ND);

    prep_starts<<<(N + 1 + 255) / 256, 256, 0, stream>>>(esrc, starts, N, E);
    prep_weights<<<(GATES * DIN + 255) / 256, 256, 0, stream>>>(
        W_ih, W_hh, b_ih, b_hh, W_out, gwih, gwhh, gwT, gbias);
    prep_xbf<<<(ND + 255) / 256, 256, 0, stream>>>(X, Xbf, ND);

    const int numTiles = (N + 15) / 16;
    int grid = numTiles < 1024 ? numTiles : 1024;
    lstm_agg_main<<<grid, 256, 0, stream>>>(
        Xbf, etrg, starts, gwih, gwhh, gwT, gbias, (float*)d_out, N, numTiles);
}
